// GraphEncoder_53592601919844
// MI455X (gfx1250) — matrix operand order repaired and checked by reading, not compiled, not run on hardware
//
#include <hip/hip_runtime.h>
#include <hip/hip_bf16.h>

typedef __attribute__((ext_vector_type(16))) __bf16 v16bf;
typedef __attribute__((ext_vector_type(8)))  __bf16 v8bf;
typedef __attribute__((ext_vector_type(8)))  float  v8f;

#define N_NODES  50000
#define N_EDGES  800000
#define DM       128
#define N_LAYER  4
#define N_GRAPHS 128
#define GCOLS    ((1 + N_LAYER) * DM)   // 640
#define CDIV(a,b) (((a)+(b)-1)/(b))

// ---------------- init: deg = 1 (self loop), g = 0 ----------------
__global__ void k_init(float* deg, unsigned* g) {
    int tid = blockIdx.x * 256 + threadIdx.x;
    if (tid < N_NODES) deg[tid] = 1.0f;
    if (tid < N_GRAPHS * GCOLS) g[tid] = 0u;
}

__global__ void k_deg(const int* __restrict__ dst, float* __restrict__ deg) {
    int e = blockIdx.x * 256 + threadIdx.x;
    if (e < N_EDGES) atomicAdd(&deg[dst[e]], 1.0f);
}

__global__ void k_dinv(const float* __restrict__ deg, float* __restrict__ dinv) {
    int i = blockIdx.x * 256 + threadIdx.x;
    if (i < N_NODES) { float d = deg[i]; dinv[i] = d > 0.f ? rsqrtf(d) : 0.f; }
}

__global__ void k_norm(const int* __restrict__ src, const int* __restrict__ dst,
                       const float* __restrict__ dinv, float* __restrict__ nrm) {
    int e = blockIdx.x * 256 + threadIdx.x;
    if (e < N_EDGES) nrm[e] = dinv[src[e]] * dinv[dst[e]];
}

// -------- encoder: u[n, 0:128] = relu(pos@Wp+bp), u[n,128:256] = relu(emb[id]) --------
__global__ void k_encode(const float* __restrict__ P, const int* __restrict__ idf,
                         const float* __restrict__ Wp, const float* __restrict__ bp,
                         const float* __restrict__ emb, __bf16* __restrict__ u) {
    int tid = blockIdx.x * 256 + threadIdx.x;
    if (tid >= N_NODES * DM) return;
    int n = tid >> 7, c = tid & (DM - 1);
    float p0 = P[n * 3 + 0], p1 = P[n * 3 + 1], p2 = P[n * 3 + 2];
    float v = fmaf(p0, Wp[c], fmaf(p1, Wp[DM + c], fmaf(p2, Wp[2 * DM + c], bp[c])));
    u[(size_t)n * (2 * DM) + c]      = (__bf16)fmaxf(v, 0.f);
    u[(size_t)n * (2 * DM) + DM + c] = (__bf16)fmaxf(emb[idf[n] * DM + c], 0.f);
}

// -------- pack f32 weights [K,128] into WMMA B-fragment order (lane%16=n, elements=k in table order) --------
__global__ void k_pack(const float* __restrict__ W, __bf16* __restrict__ P, int K) {
    int tid = blockIdx.x * 256 + threadIdx.x;
    if (tid >= K * DM) return;
    int j    = tid & 15;
    int lane = (tid >> 4) & 31;
    int nt   = (tid >> 9) & 7;
    int kc   = tid >> 12;
    int k = kc * 32 + ((lane >> 4) << 3) + (j & 7) + ((j >> 3) << 4);
    int n = nt * 16 + (lane & 15);
    P[tid] = (__bf16)W[k * DM + n];
}

// ---------------- WMMA GEMM: out = A[rows,K](bf16) @ Wpacked(K,128) ----------------
// epilogue: +bias, relu, write bf16 copy, write f32 (optionally scaled by dinv^2),
// segment-max into g via nonneg-float bit-pattern atomicMax.
__global__ __launch_bounds__(256) void k_gemm(
    const __bf16* __restrict__ A, const __bf16* __restrict__ Bp,
    const float* __restrict__ bias, const float* __restrict__ selfdinv,
    float* __restrict__ outf, __bf16* __restrict__ outbf,
    unsigned* __restrict__ gmax, const int* __restrict__ batch,
    int goff, int K, int doRelu)
{
    int lane = threadIdx.x & 31;
    int wv   = threadIdx.x >> 5;        // n-tile 0..7
    int hi   = lane >> 4;               // lane group
    int rowl = lane & 15;
    size_t row0 = (size_t)blockIdx.x * 16;
    const __bf16* arow = A + (row0 + rowl) * (size_t)K;

    v8f c = {};
    int nk = K >> 5;
    for (int kc = 0; kc < nk; ++kc) {
        int ka = (kc << 5) + (hi << 3);
        v8bf a0 = *(const v8bf*)(arow + ka);        // K = ka .. ka+7
        v8bf a1 = *(const v8bf*)(arow + ka + 16);   // K = ka+16 .. ka+23
        v16bf a = __builtin_shufflevector(a0, a1, 0,1,2,3,4,5,6,7,8,9,10,11,12,13,14,15);
        const __bf16* bpt = Bp + (((size_t)kc * 8 + wv) * 32 + lane) * 16;
        v8bf b0 = *(const v8bf*)(bpt);
        v8bf b1 = *(const v8bf*)(bpt + 8);
        v16bf b = __builtin_shufflevector(b0, b1, 0,1,2,3,4,5,6,7,8,9,10,11,12,13,14,15);
        c = __builtin_amdgcn_wmma_f32_16x16x32_bf16(
                false, a, false, b, (short)0, c, false, false);
    }

    int gcol = wv * 16 + rowl;          // C/D layout: lane%16 = column
    float bval = bias ? bias[gcol] : 0.f;
#pragma unroll
    for (int r = 0; r < 8; ++r) {
        int m = r + (hi << 3);          // lane-hi group holds rows 8..15
        size_t grow = row0 + m;
        float v = c[r] + bval;
        if (doRelu) v = fmaxf(v, 0.f);
        if (outbf) outbf[grow * DM + gcol] = (__bf16)v;
        if (outf) {
            float s = 1.f;
            if (selfdinv) { float dv = selfdinv[grow]; s = dv * dv; }
            outf[grow * DM + gcol] = s * v;
        }
        if (gmax)
            atomicMax(&gmax[(size_t)batch[grow] * GCOLS + goff + gcol],
                      __float_as_uint(v));
    }
}

// -------- edge scatter: acc[dst] += norm[e] * t[src]  (bf16 gather, f32 atomics) --------
__global__ void k_scatter(const __bf16* __restrict__ t, const int* __restrict__ src,
                          const int* __restrict__ dst, const float* __restrict__ nrm,
                          float* __restrict__ acc) {
    int tid = blockIdx.x * 256 + threadIdx.x;
    int e = tid >> 4;
    if (e >= N_EDGES) return;
    int part = tid & 15;
    int s = src[e], d = dst[e];
    float w = nrm[e];
    v8bf m = *(const v8bf*)(t + (size_t)s * DM + part * 8);
    float* ap = acc + (size_t)d * DM + part * 8;
#pragma unroll
    for (int i = 0; i < 8; ++i) atomicAdd(ap + i, w * (float)m[i]);
}

// -------- finalize layer: h = relu(acc + b); bf16 copy; segment max --------
__global__ void k_finalize(const float* __restrict__ acc, const float* __restrict__ b,
                           __bf16* __restrict__ act, unsigned* __restrict__ g,
                           const int* __restrict__ batch, int goff) {
    int tid = blockIdx.x * 256 + threadIdx.x;
    if (tid >= N_NODES * DM) return;
    int n = tid >> 7, c = tid & (DM - 1);
    float v = fmaxf(acc[tid] + b[c], 0.f);
    act[tid] = (__bf16)v;
    atomicMax(&g[(size_t)batch[n] * GCOLS + goff + c], __float_as_uint(v));
}

__global__ void k_gconv(const unsigned* __restrict__ g, __bf16* __restrict__ gbf) {
    int tid = blockIdx.x * 256 + threadIdx.x;
    if (tid < N_GRAPHS * GCOLS) gbf[tid] = (__bf16)__uint_as_float(g[tid]);
}

extern "C" void kernel_launch(void* const* d_in, const int* in_sizes, int n_in,
                              void* d_out, int out_size, void* d_ws, size_t ws_size,
                              hipStream_t stream) {
    const float* pos   = (const float*)d_in[0];
    const int*   idf   = (const int*)  d_in[1];
    const int*   srcp  = (const int*)  d_in[2];
    const int*   dstp  = srcp + N_EDGES;
    const int*   batch = (const int*)  d_in[3];
    const float* Wp    = (const float*)d_in[5];
    const float* bp    = (const float*)d_in[6];
    const float* emb   = (const float*)d_in[7];
    const float* Wn    = (const float*)d_in[8];
    const float* bn    = (const float*)d_in[9];
    const float* convW = (const float*)d_in[10];
    const float* convb = (const float*)d_in[11];
    const float* Wa    = (const float*)d_in[12];
    const float* ba    = (const float*)d_in[13];

    char* w = (char*)d_ws;
    auto alloc = [&](size_t bytes) -> void* {
        void* p = (void*)w; w += (bytes + 255) & ~(size_t)255; return p;
    };
    float*    deg   = (float*)   alloc((size_t)N_NODES * 4);
    float*    dinv  = (float*)   alloc((size_t)N_NODES * 4);
    float*    nrm   = (float*)   alloc((size_t)N_EDGES * 4);
    __bf16*   u     = (__bf16*)  alloc((size_t)N_NODES * 2 * DM * 2);
    __bf16*   act   = (__bf16*)  alloc((size_t)N_NODES * DM * 2);
    __bf16*   tbf   = (__bf16*)  alloc((size_t)N_NODES * DM * 2);
    float*    acc   = (float*)   alloc((size_t)N_NODES * DM * 4);
    unsigned* g     = (unsigned*)alloc((size_t)N_GRAPHS * GCOLS * 4);
    __bf16*   gbf   = (__bf16*)  alloc((size_t)N_GRAPHS * GCOLS * 2);
    __bf16*   pWn   = (__bf16*)  alloc((size_t)2 * DM * DM * 2);
    __bf16*   pConv = (__bf16*)  alloc((size_t)N_LAYER * DM * DM * 2);
    __bf16*   pWa   = (__bf16*)  alloc((size_t)GCOLS * DM * 2);

    // normalization
    k_init<<<CDIV(N_NODES, 256), 256, 0, stream>>>(deg, g);
    k_deg <<<CDIV(N_EDGES, 256), 256, 0, stream>>>(dstp, deg);
    k_dinv<<<CDIV(N_NODES, 256), 256, 0, stream>>>(deg, dinv);
    k_norm<<<CDIV(N_EDGES, 256), 256, 0, stream>>>(srcp, dstp, dinv, nrm);

    // weight packing (bf16, WMMA fragment order)
    k_pack<<<CDIV(2 * DM * DM, 256), 256, 0, stream>>>(Wn, pWn, 2 * DM);
    for (int l = 0; l < N_LAYER; ++l)
        k_pack<<<CDIV(DM * DM, 256), 256, 0, stream>>>(convW + (size_t)l * DM * DM,
                                                       pConv + (size_t)l * DM * DM, DM);
    k_pack<<<CDIV(GCOLS * DM, 256), 256, 0, stream>>>(Wa, pWa, GCOLS);

    // node encoder: u -> x = relu(u@Wn + bn); segment-max into g[:, 0:128]
    k_encode<<<CDIV(N_NODES * DM, 256), 256, 0, stream>>>(pos, idf, Wp, bp, emb, u);
    k_gemm<<<N_NODES / 16, 256, 0, stream>>>(u, pWn, bn, nullptr,
                                             nullptr, act, g, batch, 0, 2 * DM, 1);

    // GCN layers
    for (int l = 0; l < N_LAYER; ++l) {
        // t = h@W ; acc initialized with self-loop term dinv^2 * t
        k_gemm<<<N_NODES / 16, 256, 0, stream>>>(act, pConv + (size_t)l * DM * DM,
                                                 nullptr, dinv, acc, tbf,
                                                 nullptr, nullptr, 0, DM, 0);
        k_scatter<<<CDIV(N_EDGES * 16, 256), 256, 0, stream>>>(tbf, srcp, dstp, nrm, acc);
        k_finalize<<<CDIV(N_NODES * DM, 256), 256, 0, stream>>>(acc, convb + l * DM,
                                                                act, g, batch, (l + 1) * DM);
    }

    // readout: out = g_embed @ Wa + ba
    k_gconv<<<CDIV(N_GRAPHS * GCOLS, 256), 256, 0, stream>>>(g, gbf);
    k_gemm<<<N_GRAPHS / 16, 256, 0, stream>>>(gbf, pWa, ba, nullptr,
                                              (float*)d_out, nullptr,
                                              nullptr, nullptr, 0, GCOLS, 0);
}
